// HandDecoder_7189775253536
// MI455X (gfx1250) — matrix operand order repaired and checked by reading, not compiled, not run on hardware
//
#include <hip/hip_runtime.h>
#include <hip/hip_bf16.h>

typedef __attribute__((ext_vector_type(16))) _Float16 v16h;
typedef __attribute__((ext_vector_type(8)))  _Float16 v8h;
typedef __attribute__((ext_vector_type(8)))  float    v8f;

// Problem constants (match reference)
#define BB 32
#define NN 128
#define C0 64
#define C1 32
#define C2 16

// Workspace layout (bytes):
//   gsw  : _Float16 [B][64 ksteps][32 lanes][16 halves]  = 2 MB   @ 0
//   fbuf : float    [B][N][32]                           = 512 KB @ 2 MB
//   bias : float    [B][16]                              = 2 KB   @ 2.5 MB
#define GSW_BYTES  (2u << 20)
#define FBUF_BYTES (512u << 10)

// ---------------------------------------------------------------------------
// Kernel 1: decode MLP  f[b,j,:] = relu(mlp3(cat(feature[b], coord[b,j])))
// one thread per (b,j); 32 register accumulators, weights streamed (L2-hot)
// ---------------------------------------------------------------------------
__global__ void k_decode(const float* __restrict__ feature,
                         const float* __restrict__ coords,
                         const float* __restrict__ dw1, const float* __restrict__ db1,
                         const float* __restrict__ dw2, const float* __restrict__ db2,
                         const float* __restrict__ dw3, const float* __restrict__ db3,
                         float* __restrict__ fbuf) {
  int t = blockIdx.x * blockDim.x + threadIdx.x;   // b*N + j
  int b = t >> 7;
  int j = t & (NN - 1);

  float c0x = coords[(b * NN + j) * 3 + 0];
  float c1x = coords[(b * NN + j) * 3 + 1];
  float c2x = coords[(b * NN + j) * 3 + 2];

  // layer 1: 67 -> 32 (accumulator-major so h1 stays in registers)
  float h1[32];
#pragma unroll
  for (int o = 0; o < 32; ++o) h1[o] = db1[o];
#pragma unroll 1
  for (int i = 0; i < C0; ++i) {
    float xi = feature[b * C0 + i];
    const float* w = dw1 + i * 32;
#pragma unroll
    for (int o = 0; o < 32; ++o) h1[o] = __builtin_fmaf(xi, w[o], h1[o]);
  }
#pragma unroll
  for (int o = 0; o < 32; ++o) {
    float s = h1[o];
    s = __builtin_fmaf(c0x, dw1[64 * 32 + o], s);
    s = __builtin_fmaf(c1x, dw1[65 * 32 + o], s);
    s = __builtin_fmaf(c2x, dw1[66 * 32 + o], s);
    h1[o] = fmaxf(s, 0.f);
  }

  // layer 2: 32 -> 16 (fully unrolled; register-only)
  float h2[16];
#pragma unroll
  for (int o = 0; o < 16; ++o) {
    float s = db2[o];
#pragma unroll
    for (int i = 0; i < 32; ++i) s = __builtin_fmaf(h1[i], dw2[i * 16 + o], s);
    h2[o] = fmaxf(s, 0.f);
  }

  // layer 3: 16 -> 32, outer relu, store
#pragma unroll 1
  for (int o = 0; o < 32; ++o) {
    float s = db3[o];
#pragma unroll
    for (int k = 0; k < 16; ++k) s = __builtin_fmaf(h2[k], dw3[k * 32 + o], s);
    fbuf[t * 32 + o] = fmaxf(s, 0.f);
  }
}

// ---------------------------------------------------------------------------
// Kernel 2: G[b, j*16+k, d] = sum_c f[b,j,c] * kw3[k, c*16+d]
// stored as fp16 PRE-SWIZZLED into the WMMA B-fragment per-lane layout:
//   gsw[((b*64 + S)*32 + lane)*16 + t]  holds  G[b, 32*S + kgrp + (t&7) + (t>>3)*16, n]
//   where n = lane&15, kgrp = (lane<16 ? 0 : 8)
// one thread per (b, kflat) computing the 16 d-values of one K row.
// ---------------------------------------------------------------------------
__global__ void k_gbuild(const float* __restrict__ fbuf,
                         const float* __restrict__ kw3,
                         _Float16* __restrict__ gsw) {
  int t = blockIdx.x * blockDim.x + threadIdx.x;   // b*2048 + kflat
  int b = t >> 11;
  int kflat = t & 2047;
  int j  = kflat >> 4;
  int kh = kflat & 15;

  const float* fr = fbuf + (b * NN + j) * 32;
  int S  = kflat >> 5;
  int r  = kflat & 31;
  int tt = (r & 7) | ((r >> 4) << 3);
  int laneBase = (r & 8) ? 16 : 0;
  size_t base = ((size_t)(b * 64 + S) * 32) * 16;

#pragma unroll 1
  for (int d = 0; d < 16; ++d) {
    float s = 0.f;
#pragma unroll 8
    for (int c = 0; c < 32; ++c)
      s = __builtin_fmaf(fr[c], kw3[kh * 512 + c * 16 + d], s);
    gsw[base + (size_t)(laneBase + d) * 16 + tt] = (_Float16)s;
  }
}

// ---------------------------------------------------------------------------
// Kernel 3: bias[b,d] = sum_c (sum_j f[b,j,c]) * kb3[c*16+d]
// ---------------------------------------------------------------------------
__global__ void k_bias(const float* __restrict__ fbuf,
                       const float* __restrict__ kb3,
                       float* __restrict__ bias) {
  int b = blockIdx.x;
  int d = threadIdx.x;
  if (d >= 16) return;
  float s = 0.f;
#pragma unroll 1
  for (int c = 0; c < 32; ++c) {
    float sf = 0.f;
#pragma unroll 4
    for (int j = 0; j < NN; ++j) sf += fbuf[(b * NN + j) * 32 + c];
    s = __builtin_fmaf(sf, kb3[c * 16 + d], s);
  }
  bias[b * 16 + d] = s;
}

// ---------------------------------------------------------------------------
// Kernel 4 (main): per (b, 16-row i-tile):
//   out[b, i0+m, d] = relu( H2(b,i-tile) [16 x 2048] @ G[b] [2048 x 16] + bias )
// 4 waves/WG; wave w owns j-chunks {w, w+4}; each chunk: compute 16x(16*16)
// h2 tile (MLP 3->8->16, fp16) into its own LDS slab, then 8x WMMA f16 K-steps.
// Cross-wave partial sums reduced through LDS at the end.
// ---------------------------------------------------------------------------
__launch_bounds__(128, 1)
__global__ void k_main(const float* __restrict__ coords,
                       const float* __restrict__ kw1, const float* __restrict__ kb1,
                       const float* __restrict__ kw2, const float* __restrict__ kb2,
                       const _Float16* __restrict__ gsw,
                       const float* __restrict__ bias,
                       float* __restrict__ out) {
  __shared__ __align__(16) _Float16 sh_h2[4][16][256];  // 32 KB: per-wave h2 tile
  __shared__ float sh_w[176];                           // kw1|kb1|kw2|kb2
  __shared__ float sh_red[4][256];                      // 4 KB reduction buffer

  int b   = blockIdx.x >> 3;
  int i0  = (blockIdx.x & 7) * 16;
  int tid = threadIdx.x;
  int wave = tid >> 5;
  int lane = tid & 31;

  // stage the tiny kernel-MLP weights once into LDS
  for (int idx = tid; idx < 176; idx += 128) {
    float v;
    if      (idx < 24)  v = kw1[idx];
    else if (idx < 32)  v = kb1[idx - 24];
    else if (idx < 160) v = kw2[idx - 32];
    else                v = kb2[idx - 160];
    sh_w[idx] = v;
  }
  __syncthreads();

  const float* cb = coords + (size_t)b * NN * 3;
  const v16h*  Gs = (const v16h*)gsw;

  v8f acc = {};
  int m    = lane & 15;                 // A-fragment row
  int kb0h = (lane < 16) ? 0 : 8;       // A-fragment K sub-base

#pragma unroll 1
  for (int cc = 0; cc < 2; ++cc) {
    int chunk = wave + cc * 4;          // 0..7 → 16 j's each
    int j0 = chunk * 16;

    // ---- build h2 tile: 256 MLP evals / wave, 8 per lane ----
#pragma unroll 1
    for (int q = 0; q < 8; ++q) {
      int p  = lane + 32 * q;           // 0..255
      int il = p >> 4;
      int jl = p & 15;
      float r0 = cb[(j0 + jl) * 3 + 0] - cb[(i0 + il) * 3 + 0];
      float r1 = cb[(j0 + jl) * 3 + 1] - cb[(i0 + il) * 3 + 1];
      float r2 = cb[(j0 + jl) * 3 + 2] - cb[(i0 + il) * 3 + 2];

      float h1[8];
#pragma unroll
      for (int o = 0; o < 8; ++o) {
        float s = sh_w[24 + o];
        s = __builtin_fmaf(r0, sh_w[0 * 8 + o], s);
        s = __builtin_fmaf(r1, sh_w[1 * 8 + o], s);
        s = __builtin_fmaf(r2, sh_w[2 * 8 + o], s);
        h1[o] = fmaxf(s, 0.f);
      }
      v16h hv;
#pragma unroll
      for (int k = 0; k < 16; ++k) {
        float s = sh_w[160 + k];
#pragma unroll
        for (int o = 0; o < 8; ++o)
          s = __builtin_fmaf(h1[o], sh_w[32 + o * 16 + k], s);
        hv[k] = (_Float16)fmaxf(s, 0.f);
      }
      *(v16h*)&sh_h2[wave][il][jl * 16] = hv;   // 32B contiguous LDS store
    }
    __syncthreads();

    // ---- 8 WMMA K-steps over this chunk (K = 256) ----
    size_t gbase = (size_t)(b * 64 + chunk * 8) * 32 + lane;
#pragma unroll
    for (int s = 0; s < 8; ++s) {
      int kb = 32 * s + kb0h;
      // A fragment: elems 0..7 = halves [kb..kb+7], elems 8..15 = [kb+16..kb+23]
      v8h lo = *(const v8h*)&sh_h2[wave][m][kb];
      v8h hi = *(const v8h*)&sh_h2[wave][m][kb + 16];
      v16h a;
#pragma unroll
      for (int e = 0; e < 8; ++e) { a[e] = lo[e]; a[e + 8] = hi[e]; }

      if (s < 7) __builtin_prefetch(Gs + gbase + (size_t)(s + 1) * 32, 0, 0);
      v16h bf = Gs[gbase + (size_t)s * 32];     // pre-swizzled B fragment (32B)

      acc = __builtin_amdgcn_wmma_f32_16x16x32_f16(
          false, a, false, bf, (short)0, acc, false, false);
    }
    __syncthreads();
  }

  // ---- cross-wave reduction: C layout VGPR r → M=r+(lane<16?0:8), N=lane&15
  int mrow = (lane < 16) ? 0 : 8;
#pragma unroll
  for (int r = 0; r < 8; ++r)
    sh_red[wave][(r + mrow) * 16 + (lane & 15)] = acc[r];
  __syncthreads();

  for (int idx = tid; idx < 256; idx += 128) {
    int mi = idx >> 4;
    int d  = idx & 15;
    float s = sh_red[0][idx] + sh_red[1][idx] + sh_red[2][idx] + sh_red[3][idx]
            + bias[b * 16 + d];
    out[((size_t)(b * NN) + i0 + mi) * 16 + d] = fmaxf(s, 0.f);
  }
}

// ---------------------------------------------------------------------------
extern "C" void kernel_launch(void* const* d_in, const int* in_sizes, int n_in,
                              void* d_out, int out_size, void* d_ws, size_t ws_size,
                              hipStream_t stream) {
  const float* feature = (const float*)d_in[0];
  const float* coords  = (const float*)d_in[1];
  const float* dw1 = (const float*)d_in[2];
  const float* db1 = (const float*)d_in[3];
  const float* dw2 = (const float*)d_in[4];
  const float* db2 = (const float*)d_in[5];
  const float* dw3 = (const float*)d_in[6];
  const float* db3 = (const float*)d_in[7];
  const float* kw1 = (const float*)d_in[8];
  const float* kb1 = (const float*)d_in[9];
  const float* kw2 = (const float*)d_in[10];
  const float* kb2 = (const float*)d_in[11];
  const float* kw3 = (const float*)d_in[12];
  const float* kb3 = (const float*)d_in[13];

  char* ws = (char*)d_ws;
  _Float16* gsw = (_Float16*)ws;                              // 2 MB
  float*    fbuf = (float*)(ws + GSW_BYTES);                  // 512 KB
  float*    bias = (float*)(ws + GSW_BYTES + FBUF_BYTES);     // 2 KB
  float*    out  = (float*)d_out;

  k_decode<<<BB * NN / 128, 128, 0, stream>>>(feature, coords,
                                              dw1, db1, dw2, db2, dw3, db3, fbuf);
  k_gbuild<<<BB * NN * 16 / 256, 256, 0, stream>>>(fbuf, kw3, gsw);
  k_bias  <<<BB, 32, 0, stream>>>(fbuf, kb3, bias);
  k_main  <<<BB * (NN / 16), 128, 0, stream>>>(coords, kw1, kb1, kw2, kb2,
                                               gsw, bias, out);
}
